// AsyncLSTMAttentionMultimodal_1297080124024
// MI455X (gfx1250) — matrix operand order repaired and checked by reading, not compiled, not run on hardware
//
#include <hip/hip_runtime.h>

typedef unsigned short u16;
typedef __attribute__((ext_vector_type(16))) __bf16 v16bf;
typedef __attribute__((ext_vector_type(8)))  float  v8f;

// ---------------------------------------------------------------------------
// helpers
// ---------------------------------------------------------------------------
__device__ __forceinline__ u16 f2bf(float f) {
    union { __bf16 b; u16 u; } c; c.b = (__bf16)f; return c.u;
}
__device__ __forceinline__ float bf2f(u16 v) {
    union { __bf16 b; u16 u; } c; c.u = v; return (float)c.b;
}
__device__ __forceinline__ float sigf(float x) { return 1.0f / (1.0f + __expf(-x)); }

// A fragment: 16x32 bf16 tile from row-major [M, ld], per CDNA5 ISA 7.12.2:
//   lanes 0-15  : row = lane,    K = {k0+0..7,  k0+16..23}
//   lanes 16-31 : row = lane-16, K = {k0+8..15, k0+24..31}
__device__ __forceinline__ v16bf loadA(const u16* __restrict__ base, int ld,
                                       int row0, int k0, int lane) {
    int r = row0 + (lane & 15);
    int h = (lane >> 4) << 3;                  // 0 or 8
    const u16* p = base + (size_t)r * ld + k0 + h;
    union { v16bf v; uint4 u[2]; } f;
    f.u[0] = *(const uint4*)(p);               // K = k0+h .. k0+h+7
    f.u[1] = *(const uint4*)(p + 16);          // K = k0+16+h .. k0+16+h+7
    return f.v;
}

// B fragment: 32x16 (KxN) = W^T where W is row-major [N, ld]:
//   lanes 0-15  : col n = lane,    K = {k0+0..7,  k0+16..23}
//   lanes 16-31 : col n = lane-16, K = {k0+8..15, k0+24..31}
__device__ __forceinline__ v16bf loadB(const u16* __restrict__ base, int ld,
                                       int col0, int k0, int lane) {
    int n = col0 + (lane & 15);
    int h = (lane >> 4) << 3;                  // 0 or 8
    const u16* p = base + (size_t)n * ld + k0 + h;
    union { v16bf v; uint4 u[2]; } f;
    f.u[0] = *(const uint4*)(p);
    f.u[1] = *(const uint4*)(p + 16);
    return f.v;
}

// ---------------------------------------------------------------------------
// Generic WMMA GEMM:  D[M,N] = act( A1@W1^T (+ A2@W2^T) + (Cfull | bias | 0) )
// A row-major bf16 [M,K], W row-major bf16 [N,K].
// Each wave owns FOUR consecutive 16x16 N-tiles of one M-tile: the A fragment
// is loaded once per 32-wide K-chunk and reused for 4 v_wmma ops (2.5 b128
// loads per wmma instead of 4). 8 waves/block share the M-tile (A panel hits
// WGP$/L2). REQUIRES: N % 64 == 0 (true for every layer here) so there is no
// per-tile remainder; the only guard is a whole-wave scalar early-out, and
// EXEC stays all-1s at every WMMA as the ISA requires. The wave id is pinned
// to an SGPR via readfirstlane so all control flow is scalar branches.
// act: 0=none 1=relu 2=tanh.  outF (f32) and outB (bf16) both optional.
// ---------------------------------------------------------------------------
__global__ void gemm_wmma(const u16* __restrict__ A1, int lda1,
                          const u16* __restrict__ W1, int ldw1, int K1,
                          const u16* __restrict__ A2, int lda2,
                          const u16* __restrict__ W2, int ldw2, int K2,
                          const float* __restrict__ Cfull, int ldc,
                          const float* __restrict__ bias,
                          float* __restrict__ outF, int ldoF,
                          u16*  __restrict__ outB, int ldoB,
                          int N, int act)
{
    int lane   = threadIdx.x & 31;
    int wave   = __builtin_amdgcn_readfirstlane(threadIdx.x >> 5); // SGPR wave id
    int row0   = blockIdx.y * 16;
    int tilesN = N >> 4;
    int tile0  = blockIdx.x * 32 + wave * 4;   // 8 waves x 4 tiles per block
    if (tile0 >= tilesN) return;               // scalar branch, EXEC untouched
    int half = lane >> 4;
    int nlo  = lane & 15;

    v8f acc[4];
#pragma unroll
    for (int j = 0; j < 4; ++j) {
        int ncol = (tile0 + j) * 16 + nlo;
#pragma unroll
        for (int v = 0; v < 8; ++v) {
            int m = row0 + v + 8 * half;       // C/D layout: vgpr v, lane half
            acc[j][v] = Cfull ? Cfull[(size_t)m * ldc + ncol]
                              : (bias ? bias[ncol] : 0.0f);
        }
    }

    for (int k = 0; k < K1; k += 32) {
        if (k + 32 < K1)   // prefetch the weight stream one chunk ahead
            __builtin_prefetch(W1 + (size_t)(tile0 * 16 + nlo) * ldw1 + k + 32, 0, 3);
        v16bf a = loadA(A1, lda1, row0, k, lane);
#pragma unroll
        for (int j = 0; j < 4; ++j) {
            v16bf w = loadB(W1, ldw1, (tile0 + j) * 16, k, lane);
            acc[j] = __builtin_amdgcn_wmma_f32_16x16x32_bf16(
                         false, a, false, w, (short)0, acc[j], false, false);
        }
    }
    if (A2) {
        for (int k = 0; k < K2; k += 32) {
            v16bf a = loadA(A2, lda2, row0, k, lane);
#pragma unroll
            for (int j = 0; j < 4; ++j) {
                v16bf w = loadB(W2, ldw2, (tile0 + j) * 16, k, lane);
                acc[j] = __builtin_amdgcn_wmma_f32_16x16x32_bf16(
                             false, a, false, w, (short)0, acc[j], false, false);
            }
        }
    }

#pragma unroll
    for (int j = 0; j < 4; ++j) {
        int ncol = (tile0 + j) * 16 + nlo;
#pragma unroll
        for (int v = 0; v < 8; ++v) {
            int m = row0 + v + 8 * half;
            float x = acc[j][v];
            if (act == 1)      x = fmaxf(x, 0.0f);
            else if (act == 2) x = tanhf(x);
            if (outF) outF[(size_t)m * ldoF + ncol] = x;
            if (outB) outB[(size_t)m * ldoB + ncol] = f2bf(x);
        }
    }
}

// ---------------------------------------------------------------------------
// setup / elementwise kernels
// ---------------------------------------------------------------------------
__global__ void convpad(const float* __restrict__ src, u16* __restrict__ dst,
                        long n, int cols, int ldd)
{   // dst[r,c] = c<cols ? bf16(src[r,c]) : 0   (K-padding to mult of 32)
    long i = (long)blockIdx.x * 256 + threadIdx.x;
    if (i >= n) return;
    long r = i / ldd; int c = (int)(i % ldd);
    dst[i] = (c < cols) ? f2bf(src[(size_t)r * cols + c]) : (u16)0;
}

__global__ void addvec(const float* a, const float* b, float* o, int n)
{
    int i = blockIdx.x * 256 + threadIdx.x;
    if (i < n) o[i] = a[i] + b[i];
}

__global__ void zero_u32(unsigned* p, long n)
{
    long i = (long)blockIdx.x * 256 + threadIdx.x;
    if (i < n) p[i] = 0u;
}

// x [B,T,D] f32  ->  xT [(t*B+b), Dp] bf16 (zero padded)
__global__ void transpose_pad(const float* __restrict__ x, u16* __restrict__ xT,
                              long n, int D, int Dp)
{
    long i = (long)blockIdx.x * 256 + threadIdx.x;
    if (i >= n) return;
    long r = i / Dp; int c = (int)(i % Dp);
    int t = (int)(r >> 6), b = (int)(r & 63);   // B = 64
    xT[i] = (c < D) ? f2bf(x[((size_t)b * 256 + t) * D + c]) : (u16)0;
}

// prev_cs <- new_cs (both f32 and bf16 copies of cStar [64, 3200])
__global__ void shift_cs(float* __restrict__ cstar, u16* __restrict__ cstar_bf)
{
    int i = blockIdx.x * 256 + threadIdx.x;
    if (i >= 64 * 1600) return;
    int b = i / 1600, k = i % 1600;
    size_t d = (size_t)b * 3200 + k, s = d + 1600;
    cstar[d]    = cstar[s];
    cstar_bf[d] = cstar_bf[s];
}

// LSTM pointwise (torch gate order i,f,g,o): updates c (f32), h (bf16),
// writes c into cStar new-half (f32+bf16) and h into finalbuf row (bf16).
__global__ void lstm_point(const float* __restrict__ G, float* __restrict__ c,
                           u16* __restrict__ h_bf, float* __restrict__ cstar,
                           u16* __restrict__ cstar_bf, u16* __restrict__ frow,
                           int H, int csOff)
{
    int i = blockIdx.x * 256 + threadIdx.x;
    if (i >= 64 * H) return;
    int b = i / H, j = i % H;
    const float* g = G + (size_t)b * 4 * H;
    float ig = sigf(g[j]);
    float fg = sigf(g[H + j]);
    float gg = tanhf(g[2 * H + j]);
    float og = sigf(g[3 * H + j]);
    float c2 = fg * c[i] + ig * gg;
    float h  = og * tanhf(c2);
    c[i]    = c2;
    h_bf[i] = f2bf(h);
    size_t cs = (size_t)b * 3200 + 1600 + csOff + j;
    cstar[cs]    = c2;
    cstar_bf[cs] = f2bf(c2);
    frow[(size_t)b * 1856 + csOff + j] = f2bf(h);
}

// Attention-LSTM pointwise: updates c_att/h_att; output feature is c_att2.
__global__ void att_point(const float* __restrict__ G, float* __restrict__ c,
                          u16* __restrict__ h_bf, u16* __restrict__ frow)
{
    int i = blockIdx.x * 256 + threadIdx.x;
    if (i >= 64 * 256) return;
    int b = i >> 8, j = i & 255;
    const float* g = G + (size_t)b * 1024;
    float ig = sigf(g[j]);
    float fg = sigf(g[256 + j]);
    float gg = tanhf(g[512 + j]);
    float og = sigf(g[768 + j]);
    float c2 = fg * c[i] + ig * gg;
    float h  = og * tanhf(c2);
    c[i]    = c2;
    h_bf[i] = f2bf(h);
    frow[(size_t)b * 1856 + 1600 + j] = f2bf(c2);
}

// softmax over 3200 features per batch row, then attended = att * cStar (bf16)
__global__ void softmax_attend(const float* __restrict__ logits,
                               const float* __restrict__ cstar,
                               u16* __restrict__ attended)
{
    const int N = 3200;
    int b = blockIdx.x, tid = threadIdx.x;
    __shared__ float red[8];
    __shared__ float bcast;
    const float* L = logits + (size_t)b * N;

    float m = -3.0e38f;
    for (int k = tid; k < N; k += 256) m = fmaxf(m, L[k]);
    for (int o = 16; o > 0; o >>= 1) m = fmaxf(m, __shfl_down(m, o, 32));
    if ((tid & 31) == 0) red[tid >> 5] = m;
    __syncthreads();
    if (tid == 0) { float mm = red[0]; for (int i = 1; i < 8; ++i) mm = fmaxf(mm, red[i]); bcast = mm; }
    __syncthreads();
    m = bcast;

    float s = 0.0f;
    for (int k = tid; k < N; k += 256) s += __expf(L[k] - m);
    for (int o = 16; o > 0; o >>= 1) s += __shfl_down(s, o, 32);
    if ((tid & 31) == 0) red[tid >> 5] = s;
    __syncthreads();
    if (tid == 0) { float ss = 0.0f; for (int i = 0; i < 8; ++i) ss += red[i]; bcast = ss; }
    __syncthreads();
    float inv = 1.0f / bcast;

    for (int k = tid; k < N; k += 256) {
        size_t idx = (size_t)b * N + k;
        attended[idx] = f2bf(__expf(L[k] - m) * inv * cstar[idx]);
    }
}

// final N=1 layer: one wave per row; out[b*T+t] = (H1[r,:].w2 + b2) * mask
__global__ void out_dot(const u16* __restrict__ H1, const float* __restrict__ w2,
                        const float* __restrict__ b2, const float* __restrict__ mask,
                        float* __restrict__ out)
{
    int row  = (blockIdx.x * 256 + threadIdx.x) >> 5;   // r = t*64 + b
    int lane = threadIdx.x & 31;
    if (row >= 16384) return;
    float s = 0.0f;
    for (int k = lane; k < 256; k += 32) s += bf2f(H1[(size_t)row * 256 + k]) * w2[k];
    for (int o = 16; o > 0; o >>= 1) s += __shfl_down(s, o, 32);
    if (lane == 0) {
        int b = row & 63, t = row >> 6;
        out[b * 256 + t] = (s + b2[0]) * mask[b * 256 + t];
    }
}

// ---------------------------------------------------------------------------
// host orchestration
// ---------------------------------------------------------------------------
extern "C" void kernel_launch(void* const* d_in, const int* in_sizes, int n_in,
                              void* d_out, int out_size, void* d_ws, size_t ws_size,
                              hipStream_t stream)
{
    (void)in_sizes; (void)n_in; (void)out_size; (void)ws_size;
    const float* x_l   = (const float*)d_in[0];
    const float* x_a   = (const float*)d_in[1];
    const float* x_i   = (const float*)d_in[2];
    const float* mask  = (const float*)d_in[3];
    const float* Wih_l = (const float*)d_in[4];
    const float* Whh_l = (const float*)d_in[5];
    const float* bih_l = (const float*)d_in[6];
    const float* bhh_l = (const float*)d_in[7];
    const float* Wih_a = (const float*)d_in[8];
    const float* Whh_a = (const float*)d_in[9];
    const float* bih_a = (const float*)d_in[10];
    const float* bhh_a = (const float*)d_in[11];
    const float* Wih_i = (const float*)d_in[12];
    const float* Whh_i = (const float*)d_in[13];
    const float* bih_i = (const float*)d_in[14];
    const float* bhh_i = (const float*)d_in[15];
    const float* a1w1  = (const float*)d_in[16];
    const float* a1b1  = (const float*)d_in[17];
    const float* a1w2  = (const float*)d_in[18];
    const float* a1b2  = (const float*)d_in[19];
    const float* a2w1  = (const float*)d_in[20];
    const float* a2b1  = (const float*)d_in[21];
    const float* a2w2  = (const float*)d_in[22];
    const float* a2b2  = (const float*)d_in[23];
    const float* Wih_t = (const float*)d_in[24];
    const float* Whh_t = (const float*)d_in[25];
    const float* bih_t = (const float*)d_in[26];
    const float* bhh_t = (const float*)d_in[27];
    const float* ow1   = (const float*)d_in[28];
    const float* ob1   = (const float*)d_in[29];
    const float* ow2   = (const float*)d_in[30];
    const float* ob2   = (const float*)d_in[31];

    // ---- workspace allocator (256B aligned) ----
    size_t off = 0;
    auto alloc = [&](size_t bytes) -> void* {
        void* p = (char*)d_ws + off;
        off += (bytes + 255) & ~(size_t)255;
        return p;
    };

    // bf16 weights (K padded to mult of 32): all live in 192MB L2 across steps
    u16* Wihl_bf = (u16*)alloc(2048UL * 320 * 2);
    u16* Whhl_bf = (u16*)alloc(2048UL * 512 * 2);
    u16* Wiha_bf = (u16*)alloc(256UL  * 96  * 2);
    u16* Whha_bf = (u16*)alloc(256UL  * 64  * 2);
    u16* Wihi_bf = (u16*)alloc(4096UL * 2048 * 2);
    u16* Whhi_bf = (u16*)alloc(4096UL * 1024 * 2);
    u16* a1w1_bf = (u16*)alloc(128UL  * 3200 * 2);
    u16* a1w2_bf = (u16*)alloc(3200UL * 128 * 2);
    u16* a2w1_bf = (u16*)alloc(256UL  * 3200 * 2);
    u16* a2w2_bf = (u16*)alloc(1600UL * 256 * 2);
    u16* Wiht_bf = (u16*)alloc(1024UL * 1600 * 2);
    u16* Whht_bf = (u16*)alloc(1024UL * 256 * 2);
    u16* ow1_bf  = (u16*)alloc(256UL  * 1856 * 2);
    float* bcl = (float*)alloc(2048 * 4);
    float* bca = (float*)alloc(256 * 4);
    float* bci = (float*)alloc(4096 * 4);
    float* bct = (float*)alloc(1024 * 4);
    // time-major bf16 inputs + precomputed f32 input projections
    u16*   xTl = (u16*)alloc(16384UL * 320 * 2);
    u16*   xTa = (u16*)alloc(16384UL * 96 * 2);
    u16*   xTi = (u16*)alloc(16384UL * 2048 * 2);
    float* Xpl = (float*)alloc(16384UL * 2048 * 4);
    float* Xpa = (float*)alloc(16384UL * 256 * 4);
    float* Xpi = (float*)alloc(16384UL * 4096 * 4);
    // concat feature buffer [t*B+b, 1600 hs | 256 c_att] bf16, and MLP hidden
    u16* finalbuf = (u16*)alloc(16384UL * 1856 * 2);
    u16* H1       = (u16*)alloc(16384UL * 256 * 2);
    // per-step scratch
    float* Gl = (float*)alloc(64UL * 2048 * 4);
    float* Ga = (float*)alloc(64UL * 256 * 4);
    float* Gi = (float*)alloc(64UL * 4096 * 4);
    float* Gt = (float*)alloc(64UL * 1024 * 4);
    float* logits = (float*)alloc(64UL * 3200 * 4);
    u16* t128     = (u16*)alloc(64UL * 128 * 2);
    u16* attended = (u16*)alloc(64UL * 3200 * 2);
    u16* t256     = (u16*)alloc(64UL * 256 * 2);
    u16* cHat     = (u16*)alloc(64UL * 1600 * 2);
    // recurrent state (zero-initialized every call -> deterministic)
    char* state0 = (char*)d_ws + off;
    float* cl = (float*)alloc(64UL * 512 * 4);
    float* ca = (float*)alloc(64UL * 64 * 4);
    float* ci = (float*)alloc(64UL * 1024 * 4);
    float* ct = (float*)alloc(64UL * 256 * 4);
    float* cstar = (float*)alloc(64UL * 3200 * 4);
    u16* hl = (u16*)alloc(64UL * 512 * 2);
    u16* ha = (u16*)alloc(64UL * 64 * 2);
    u16* hi = (u16*)alloc(64UL * 1024 * 2);
    u16* ht = (u16*)alloc(64UL * 256 * 2);
    u16* cstar_bf = (u16*)alloc(64UL * 3200 * 2);
    size_t stateBytes = (size_t)(((char*)d_ws + off) - state0);

    auto cdiv = [](long a, long b) -> int { return (int)((a + b - 1) / b); };

    auto gemm = [&](const u16* A1, int lda1, const u16* W1, int ldw1, int K1,
                    const u16* A2, int lda2, const u16* W2, int ldw2, int K2,
                    const float* Cf, int ldc, const float* bias,
                    float* oF, int ldoF, u16* oB, int ldoB,
                    int M, int N, int act) {
        int tilesN = N / 16;                    // always a multiple of 4 here
        dim3 grid((unsigned)((tilesN + 31) / 32), (unsigned)(M / 16));
        gemm_wmma<<<grid, 256, 0, stream>>>(A1, lda1, W1, ldw1, K1,
                                            A2, lda2, W2, ldw2, K2,
                                            Cf, ldc, bias, oF, ldoF, oB, ldoB,
                                            N, act);
    };

    // ---- one-time setup: weight/bias conversion, input transpose ----
    convpad<<<cdiv(2048L*320,256),256,0,stream>>>(Wih_l, Wihl_bf, 2048L*320, 300, 320);
    convpad<<<cdiv(2048L*512,256),256,0,stream>>>(Whh_l, Whhl_bf, 2048L*512, 512, 512);
    convpad<<<cdiv(256L*96,256),256,0,stream>>>(Wih_a, Wiha_bf, 256L*96, 74, 96);
    convpad<<<cdiv(256L*64,256),256,0,stream>>>(Whh_a, Whha_bf, 256L*64, 64, 64);
    convpad<<<cdiv(4096L*2048,256),256,0,stream>>>(Wih_i, Wihi_bf, 4096L*2048, 2048, 2048);
    convpad<<<cdiv(4096L*1024,256),256,0,stream>>>(Whh_i, Whhi_bf, 4096L*1024, 1024, 1024);
    convpad<<<cdiv(128L*3200,256),256,0,stream>>>(a1w1, a1w1_bf, 128L*3200, 3200, 3200);
    convpad<<<cdiv(3200L*128,256),256,0,stream>>>(a1w2, a1w2_bf, 3200L*128, 128, 128);
    convpad<<<cdiv(256L*3200,256),256,0,stream>>>(a2w1, a2w1_bf, 256L*3200, 3200, 3200);
    convpad<<<cdiv(1600L*256,256),256,0,stream>>>(a2w2, a2w2_bf, 1600L*256, 256, 256);
    convpad<<<cdiv(1024L*1600,256),256,0,stream>>>(Wih_t, Wiht_bf, 1024L*1600, 1600, 1600);
    convpad<<<cdiv(1024L*256,256),256,0,stream>>>(Whh_t, Whht_bf, 1024L*256, 256, 256);
    convpad<<<cdiv(256L*1856,256),256,0,stream>>>(ow1, ow1_bf, 256L*1856, 1856, 1856);
    addvec<<<cdiv(2048,256),256,0,stream>>>(bih_l, bhh_l, bcl, 2048);
    addvec<<<1,256,0,stream>>>(bih_a, bhh_a, bca, 256);
    addvec<<<cdiv(4096,256),256,0,stream>>>(bih_i, bhh_i, bci, 4096);
    addvec<<<cdiv(1024,256),256,0,stream>>>(bih_t, bhh_t, bct, 1024);
    transpose_pad<<<cdiv(16384L*320,256),256,0,stream>>>(x_l, xTl, 16384L*320, 300, 320);
    transpose_pad<<<cdiv(16384L*96,256),256,0,stream>>>(x_a, xTa, 16384L*96, 74, 96);
    transpose_pad<<<cdiv(16384L*2048,256),256,0,stream>>>(x_i, xTi, 16384L*2048, 2048, 2048);
    zero_u32<<<cdiv((long)(stateBytes/4),256),256,0,stream>>>((unsigned*)state0, (long)(stateBytes/4));

    // ---- hoisted input projections (big parallel WMMA GEMMs) ----
    gemm(xTl, 320, Wihl_bf, 320, 320, nullptr,0,nullptr,0,0,
         nullptr,0, bcl, Xpl, 2048, nullptr,0, 16384, 2048, 0);
    gemm(xTa, 96, Wiha_bf, 96, 96, nullptr,0,nullptr,0,0,
         nullptr,0, bca, Xpa, 256, nullptr,0, 16384, 256, 0);
    gemm(xTi, 2048, Wihi_bf, 2048, 2048, nullptr,0,nullptr,0,0,
         nullptr,0, bci, Xpi, 4096, nullptr,0, 16384, 4096, 0);

    // ---- sequential scan over T=256 steps ----
    for (int t = 0; t < 256; ++t) {
        shift_cs<<<cdiv(64L*1600,256),256,0,stream>>>(cstar, cstar_bf);

        // recurrent gate GEMMs: C-operand initialized from precomputed Xp
        gemm(hl, 512, Whhl_bf, 512, 512, nullptr,0,nullptr,0,0,
             Xpl + (size_t)t*64*2048, 2048, nullptr, Gl, 2048, nullptr,0, 64, 2048, 0);
        gemm(ha, 64, Whha_bf, 64, 64, nullptr,0,nullptr,0,0,
             Xpa + (size_t)t*64*256, 256, nullptr, Ga, 256, nullptr,0, 64, 256, 0);
        gemm(hi, 1024, Whhi_bf, 1024, 1024, nullptr,0,nullptr,0,0,
             Xpi + (size_t)t*64*4096, 4096, nullptr, Gi, 4096, nullptr,0, 64, 4096, 0);

        u16* frow = finalbuf + (size_t)t * 64 * 1856;
        lstm_point<<<cdiv(64L*512,256),256,0,stream>>>(Gl, cl, hl, cstar, cstar_bf, frow, 512, 0);
        lstm_point<<<cdiv(64L*64,256),256,0,stream>>>(Ga, ca, ha, cstar, cstar_bf, frow, 64, 512);
        lstm_point<<<cdiv(64L*1024,256),256,0,stream>>>(Gi, ci, hi, cstar, cstar_bf, frow, 1024, 576);

        // attention: relu(cStar@w1^T+b1)@w2^T+b2 -> softmax(features) * cStar
        gemm(cstar_bf, 3200, a1w1_bf, 3200, 3200, nullptr,0,nullptr,0,0,
             nullptr,0, a1b1, nullptr,0, t128, 128, 64, 128, 1);
        gemm(t128, 128, a1w2_bf, 128, 128, nullptr,0,nullptr,0,0,
             nullptr,0, a1b2, logits, 3200, nullptr,0, 64, 3200, 0);
        softmax_attend<<<64,256,0,stream>>>(logits, cstar, attended);
        gemm(attended, 3200, a2w1_bf, 3200, 3200, nullptr,0,nullptr,0,0,
             nullptr,0, a2b1, nullptr,0, t256, 256, 64, 256, 1);
        gemm(t256, 256, a2w2_bf, 256, 256, nullptr,0,nullptr,0,0,
             nullptr,0, a2b2, nullptr,0, cHat, 1600, 64, 1600, 2);

        // attention LSTM: gates = cHat@Wih^T + h_att@Whh^T + (bih+bhh)
        gemm(cHat, 1600, Wiht_bf, 1600, 1600, ht, 256, Whht_bf, 256, 256,
             nullptr,0, bct, Gt, 1024, nullptr,0, 64, 1024, 0);
        att_point<<<cdiv(64L*256,256),256,0,stream>>>(Gt, ct, ht, frow);
    }

    // ---- output MLP over all (t,b): relu(final@ow1^T+b1)@ow2^T+b2, masked ----
    gemm(finalbuf, 1856, ow1_bf, 1856, 1856, nullptr,0,nullptr,0,0,
         nullptr,0, ob1, nullptr,0, H1, 256, 16384, 256, 1);
    out_dot<<<2048,256,0,stream>>>(H1, ow2, ob2, mask, (float*)d_out);
}
